// CentroidLayer_26603027431729
// MI455X (gfx1250) — matrix operand order repaired and checked by reading, not compiled, not run on hardware
//
#include <hip/hip_runtime.h>
#include <cstddef>

// ---------------------------------------------------------------------------
// CentroidLayer on MI455X (gfx1250, wave32, WMMA)
//
//   d2 = |x|^2 - 2 x@c^T + |c|^2   via bf16x3 split-GEMM on v_wmma_f32_16x16x32_bf16
//   dist = sqrt(max(d2,0)); per-class min over 4 centroids; sigmoid accept gate.
//
// GEMM inner loop: LDS double-buffered, register-prefetched, 1 barrier/K-step.
// ---------------------------------------------------------------------------

typedef __bf16 bf16_t;
typedef __attribute__((ext_vector_type(16))) __bf16 v16bf;
typedef __attribute__((ext_vector_type(8)))  __bf16 v8bf;
typedef __attribute__((ext_vector_type(4)))  __bf16 v4bf;
typedef __attribute__((ext_vector_type(8)))  float  v8f;

namespace {
constexpr int kD     = 1024;   // feature dim
constexpr int kB     = 4096;   // batch rows
constexpr int kNReal = 4000;   // C*N real centroid rows
constexpr int kNPad  = 4096;   // centroid rows padded to tile multiple
constexpr int kClass = 1000;   // classes
constexpr int kOutC  = 1001;   // output columns (classes + accept)

constexpr int BLK_M = 128;
constexpr int BLK_N = 128;
constexpr int BLK_K = 32;
constexpr int KT    = kD / BLK_K;   // 32 K-steps
constexpr int LDSS  = 40;      // halves per LDS row: 32 data + 8 pad (16B align, bank-spread)
} // namespace

// ---------------------------------------------------------------------------
// fp32 -> (bf16 hi, bf16 lo) split + exact row squared-norm.
// One block per row, 256 threads, 4 contiguous elements per thread.
// Rows >= nreal are zero-padded (their outputs are never stored).
// ---------------------------------------------------------------------------
__global__ __launch_bounds__(256)
void prep_split_kernel(const float* __restrict__ src, bf16_t* __restrict__ hi,
                       bf16_t* __restrict__ lo, float* __restrict__ sq, int nreal) {
  const int row = blockIdx.x;
  const int tid = threadIdx.x;
  const int i   = tid * 4;                  // 256 threads * 4 = 1024 = kD
  bf16_t* hp = hi + (size_t)row * kD + i;
  bf16_t* lp = lo + (size_t)row * kD + i;

  float ssum = 0.0f;
  if (row < nreal) {
    const float4 f = *(const float4*)(src + (size_t)row * kD + i);
    const bf16_t h0 = (bf16_t)f.x, h1 = (bf16_t)f.y, h2 = (bf16_t)f.z, h3 = (bf16_t)f.w;
    const v4bf vh = {h0, h1, h2, h3};
    const v4bf vl = {(bf16_t)(f.x - (float)h0), (bf16_t)(f.y - (float)h1),
                     (bf16_t)(f.z - (float)h2), (bf16_t)(f.w - (float)h3)};
    *(v4bf*)hp = vh;
    *(v4bf*)lp = vl;
    ssum = f.x * f.x + f.y * f.y + f.z * f.z + f.w * f.w;
  } else {
    const v4bf z = {(bf16_t)0.0f, (bf16_t)0.0f, (bf16_t)0.0f, (bf16_t)0.0f};
    *(v4bf*)hp = z;
    *(v4bf*)lp = z;
  }

  for (int off = 16; off > 0; off >>= 1) ssum += __shfl_down(ssum, off, 32);
  __shared__ float red[8];
  if ((tid & 31) == 0) red[tid >> 5] = ssum;
  __syncthreads();
  if (tid == 0) {
    float t = 0.0f;
    for (int w = 0; w < 8; ++w) t += red[w];
    sq[row] = t;
  }
}

// ---------------------------------------------------------------------------
// Tiled bf16x3 GEMM + distance epilogue.
// Block = 256 threads (8 waves) -> 128x128 output tile.
// Wave (wm in 0..3, wn in 0..1) owns a 32x64 sub-tile = 2x4 WMMA accumulators.
// Per K-step of 32: acc += Ahi*Bhi + Ahi*Blo + Alo*Bhi  (fp32 accumulate).
// ---------------------------------------------------------------------------
__global__ __launch_bounds__(256)
void centroid_gemm_kernel(const bf16_t* __restrict__ xhi, const bf16_t* __restrict__ xlo,
                          const bf16_t* __restrict__ chi, const bf16_t* __restrict__ clo,
                          const float* __restrict__ x2,  const float* __restrict__ c2,
                          float* __restrict__ out) {
  // double-buffered stages: 2 * 4 * 128*40*2B = 80 KB
  __shared__ bf16_t sAh[2][BLK_M * LDSS];
  __shared__ bf16_t sAl[2][BLK_M * LDSS];
  __shared__ bf16_t sBh[2][BLK_N * LDSS];
  __shared__ bf16_t sBl[2][BLK_N * LDSS];

  const int tid  = threadIdx.x;
  const int lane = tid & 31;
  const int wave = tid >> 5;
  const int wm   = wave & 3;   // 4 waves along M (32 rows each)
  const int wn   = wave >> 2;  // 2 waves along N (64 cols each)

  const int m0 = blockIdx.x * BLK_M;
  const int n0 = blockIdx.y * BLK_N;

  v8f acc[2][4];
  const v8f vzero = {0.f, 0.f, 0.f, 0.f, 0.f, 0.f, 0.f, 0.f};
  for (int i = 0; i < 2; ++i)
    for (int j = 0; j < 4; ++j) acc[i][j] = vzero;

  // Staging geometry: 512 chunks of 8 halves per tile; 2 chunks per thread.
  const int chunkR[2]  = {(tid * 2 + 0) >> 2, (tid * 2 + 1) >> 2};      // 0..127
  const int chunkKo[2] = {((tid * 2 + 0) & 3) * 8, ((tid * 2 + 1) & 3) * 8};

  // WMMA operand lane mapping used here (16-bit A 16x32 / B 32x16, wave32): contiguous k, not the ISA table order
  const int frow = lane & 15;           // M (for A) or N (for B)
  const int kbA  = (lane >> 4) * 16;    // A: lanes 16-31 hold K=16..31
  const int kbB  = (lane >> 4) * 16;    // B: lanes 16-31 hold K=16..31

  // ---- prologue: stage K-tile 0 into buffer 0 (direct global -> LDS) ----
  {
#pragma unroll
    for (int c = 0; c < 2; ++c) {
      const int r = chunkR[c], ko = chunkKo[c];
      const size_t gA = (size_t)(m0 + r) * kD + ko;
      const size_t gB = (size_t)(n0 + r) * kD + ko;
      *(v8bf*)&sAh[0][r * LDSS + ko] = *(const v8bf*)&xhi[gA];
      *(v8bf*)&sAl[0][r * LDSS + ko] = *(const v8bf*)&xlo[gA];
      *(v8bf*)&sBh[0][r * LDSS + ko] = *(const v8bf*)&chi[gB];
      *(v8bf*)&sBl[0][r * LDSS + ko] = *(const v8bf*)&clo[gB];
    }
  }
  __syncthreads();

  for (int kt = 0; kt < KT; ++kt) {
    const int buf  = kt & 1;
    const bool next = (kt + 1 < KT);

    // ---- 1) issue next tile's global loads into registers (no wait here) ----
    v8bf rAh[2], rAl[2], rBh[2], rBl[2];
    if (next) {
      const int k0n = (kt + 1) * BLK_K;
#pragma unroll
      for (int c = 0; c < 2; ++c) {
        const int r = chunkR[c], ko = chunkKo[c];
        const size_t gA = (size_t)(m0 + r) * kD + k0n + ko;
        const size_t gB = (size_t)(n0 + r) * kD + k0n + ko;
        rAh[c] = *(const v8bf*)&xhi[gA];
        rAl[c] = *(const v8bf*)&xlo[gA];
        rBh[c] = *(const v8bf*)&chi[gB];
        rBl[c] = *(const v8bf*)&clo[gB];
        if (c == 0 && kt + 2 < KT) {     // stream 2 K-steps ahead into cache
          __builtin_prefetch(&xhi[gA + BLK_K], 0, 3);
          __builtin_prefetch(&xlo[gA + BLK_K], 0, 3);
          __builtin_prefetch(&chi[gB + BLK_K], 0, 3);
          __builtin_prefetch(&clo[gB + BLK_K], 0, 3);
        }
      }
    }

    // ---- 2) compute from current LDS stage ----
    v16bf ah[2], al[2];
#pragma unroll
    for (int i = 0; i < 2; ++i) {
      const int mrow = wm * 32 + i * 16 + frow;
      const bf16_t* p = &sAh[buf][mrow * LDSS + kbA];
      const v8bf p0 = *(const v8bf*)p;           // K = kbA .. kbA+7
      const v8bf p1 = *(const v8bf*)(p + 8);     // K = kbA+8 .. kbA+15
      ah[i] = __builtin_shufflevector(p0, p1, 0,1,2,3,4,5,6,7,8,9,10,11,12,13,14,15);
      const bf16_t* q = &sAl[buf][mrow * LDSS + kbA];
      const v8bf q0 = *(const v8bf*)q;
      const v8bf q1 = *(const v8bf*)(q + 8);
      al[i] = __builtin_shufflevector(q0, q1, 0,1,2,3,4,5,6,7,8,9,10,11,12,13,14,15);
    }
#pragma unroll
    for (int j = 0; j < 4; ++j) {
      const int nrow = wn * 64 + j * 16 + frow;
      const bf16_t* p = &sBh[buf][nrow * LDSS + kbB];
      const v8bf b0 = *(const v8bf*)p;           // K = kbB .. kbB+7
      const v8bf b1 = *(const v8bf*)(p + 8);     // K = kbB+8 .. kbB+15
      const v16bf bh = __builtin_shufflevector(b0, b1, 0,1,2,3,4,5,6,7,8,9,10,11,12,13,14,15);
      const bf16_t* q = &sBl[buf][nrow * LDSS + kbB];
      const v8bf c0 = *(const v8bf*)q;
      const v8bf c1 = *(const v8bf*)(q + 8);
      const v16bf bl = __builtin_shufflevector(c0, c1, 0,1,2,3,4,5,6,7,8,9,10,11,12,13,14,15);
#pragma unroll
      for (int i = 0; i < 2; ++i) {
        acc[i][j] = __builtin_amdgcn_wmma_f32_16x16x32_bf16(
            false, ah[i], false, bh, (short)0, acc[i][j], false, false);
        acc[i][j] = __builtin_amdgcn_wmma_f32_16x16x32_bf16(
            false, ah[i], false, bl, (short)0, acc[i][j], false, false);
        acc[i][j] = __builtin_amdgcn_wmma_f32_16x16x32_bf16(
            false, al[i], false, bh, (short)0, acc[i][j], false, false);
      }
    }

    // ---- 3) drain prefetched registers into the other LDS stage ----
    if (next) {
      const int nb = buf ^ 1;
#pragma unroll
      for (int c = 0; c < 2; ++c) {
        const int r = chunkR[c], ko = chunkKo[c];
        *(v8bf*)&sAh[nb][r * LDSS + ko] = rAh[c];
        *(v8bf*)&sAl[nb][r * LDSS + ko] = rAl[c];
        *(v8bf*)&sBh[nb][r * LDSS + ko] = rBh[c];
        *(v8bf*)&sBl[nb][r * LDSS + ko] = rBl[c];
      }
    }
    __syncthreads();   // single barrier per K-step
  }

  // ---- epilogue: d2 -> dist -> per-class min over 4 centroids -> y = -dist ----
  // C/D layout: lane L holds N = L&15; VGPR r holds M = r + 8*(L>>4).
  const int nlane = lane & 15;
  const int mhalf = (lane >> 4) * 8;
#pragma unroll
  for (int i = 0; i < 2; ++i) {
    float xs[8];
#pragma unroll
    for (int r = 0; r < 8; ++r)
      xs[r] = x2[m0 + wm * 32 + i * 16 + mhalf + r];
#pragma unroll
    for (int j = 0; j < 4; ++j) {
      const int n  = n0 + wn * 64 + j * 16 + nlane;
      const float cc = c2[n];
#pragma unroll
      for (int r = 0; r < 8; ++r) {
        const float d2   = xs[r] - 2.0f * acc[i][j][r] + cc;
        const float dist = sqrtf(fmaxf(d2, 0.0f));
        // 4 consecutive lanes = 4 centroids of one class (stays within each 16-lane half)
        float dmin = fminf(dist, __shfl_xor(dist, 1, 32));
        dmin       = fminf(dmin, __shfl_xor(dmin, 2, 32));
        if ((lane & 3) == 0 && n < kNReal) {
          const int m = m0 + wm * 32 + i * 16 + mhalf + r;
          out[(size_t)m * kOutC + (n >> 2)] = -dmin;
        }
      }
    }
  }
}

// ---------------------------------------------------------------------------
// Rejection head: min_dist = -max_c y[b,c]; sigmoid gate into column 1000.
// ---------------------------------------------------------------------------
__global__ __launch_bounds__(256)
void accept_kernel(const float* __restrict__ y,
                   const float* __restrict__ std_scale, const float* __restrict__ ac_temp,
                   const float* __restrict__ running_mean, const float* __restrict__ running_var,
                   float* __restrict__ out) {
  const int row = blockIdx.x;
  const int tid = threadIdx.x;
  const float* yr = y + (size_t)row * kOutC;

  float mx = -3.402823466e38f;
  for (int i = tid; i < kClass; i += 256) mx = fmaxf(mx, yr[i]);
  for (int off = 16; off > 0; off >>= 1) mx = fmaxf(mx, __shfl_xor(mx, off, 32));
  __shared__ float red[8];
  if ((tid & 31) == 0) red[tid >> 5] = mx;
  __syncthreads();
  if (tid == 0) {
    float m = red[0];
    for (int w = 1; w < 8; ++w) m = fmaxf(m, red[w]);
    const float min_dist = -m;
    float ss = fmaxf(std_scale[0], 0.0f);
    ss = fminf(ss, 5.0f);                                   // clip(relu(std_scale),0,AC_STD_LIM)
    const float max_ac = running_mean[0] + ss * sqrtf(running_var[0]);
    const float z = (max_ac - min_dist) / ac_temp[0];
    out[(size_t)row * kOutC + kClass] = 1.0f / (1.0f + __expf(-z));
  }
}

// ---------------------------------------------------------------------------
// kernel_launch
// Workspace layout (needs ~32.03 MB):
//   xhi[4096*1024] bf16 | xlo | chi[4096*1024] bf16 (rows>=4000 zero) | clo |
//   x2[4096] f32 | c2[4096] f32
// ---------------------------------------------------------------------------
extern "C" void kernel_launch(void* const* d_in, const int* in_sizes, int n_in,
                              void* d_out, int out_size, void* d_ws, size_t ws_size,
                              hipStream_t stream) {
  (void)in_sizes; (void)n_in; (void)out_size; (void)ws_size;

  const float* x            = (const float*)d_in[0];
  const float* centroids    = (const float*)d_in[1];
  const float* std_scale    = (const float*)d_in[2];
  const float* ac_temp      = (const float*)d_in[3];
  const float* running_mean = (const float*)d_in[4];
  const float* running_var  = (const float*)d_in[5];
  float* out = (float*)d_out;

  bf16_t* xhi = (bf16_t*)d_ws;
  bf16_t* xlo = xhi + (size_t)kB * kD;
  bf16_t* chi = xlo + (size_t)kB * kD;
  bf16_t* clo = chi + (size_t)kNPad * kD;
  float*  x2  = (float*)(clo + (size_t)kNPad * kD);
  float*  c2  = x2 + kB;

  prep_split_kernel<<<kB,    256, 0, stream>>>(x,         xhi, xlo, x2, kB);
  prep_split_kernel<<<kNPad, 256, 0, stream>>>(centroids, chi, clo, c2, kNReal);

  dim3 grid(kB / BLK_M, kNPad / BLK_N);
  centroid_gemm_kernel<<<grid, 256, 0, stream>>>(xhi, xlo, chi, clo, x2, c2, out);

  accept_kernel<<<kB, 256, 0, stream>>>(out, std_scale, ac_temp, running_mean,
                                        running_var, out);
}
